// MoE_68607807586392
// MI455X (gfx1250) — compile-verified, matrix operand order checked by reading, not run on hardware
//
#include <hip/hip_runtime.h>
#include <hip/hip_bf16.h>

// ---------------------------------------------------------------------------
// MoE forward for MI455X (gfx1250, wave32, WMMA).
// Routed (token,expert) pairs only: N*k = 16384 FFN rows, ~275 GFLOP.
// bf16 WMMA (v_wmma_f32_16x16x32_bf16) with f32 accumulation; bf16 weights
// (128 MB) fit in the 192 MB L2 so cross-tile weight rereads are L2 hits.
// Tiles are staged with GLOBAL_LOAD_ASYNC_TO_LDS_B128 (ASYNCcnt) into
// double-buffered LDS: one barrier per 64-wide K step, DMA overlapped with WMMA.
// ---------------------------------------------------------------------------

#define N_TOK 4096
#define DIM   1024
#define FFN   4096
#define NEXP  8
#define TOPK  4
#define KSTEP 64
#define LDSW  72          // padded LDS row stride (bf16): 36 banks -> conflict-free

typedef __attribute__((ext_vector_type(16))) __bf16       v16bf;
typedef __attribute__((ext_vector_type(8)))  float        v8f;
typedef __attribute__((ext_vector_type(4)))  unsigned int u32x4;

// Async-LDS builtin takes pointers to 4xi32 vectors in AS1 (global) / AS3 (LDS).
typedef int i32x4v __attribute__((vector_size(16)));
typedef __attribute__((address_space(1))) i32x4v gvec128;
typedef __attribute__((address_space(3))) i32x4v lvec128;

union FragAB { u32x4 q[2]; v16bf f; };
union Vec8   { u32x4 q;    unsigned short s[8]; };

#if defined(__has_builtin)
#  if __has_builtin(__builtin_amdgcn_global_load_async_to_lds_b128)
#    define HAVE_ASYNC_LDS 1
#  endif
#endif
#ifndef HAVE_ASYNC_LDS
#  define HAVE_ASYNC_LDS 0
#endif

// 16B global -> LDS copy; async DMA path on CDNA5 (no VGPR round trip).
__device__ __forceinline__ void g2l_b128(const unsigned short* g, unsigned short* l) {
#if HAVE_ASYNC_LDS
    __builtin_amdgcn_global_load_async_to_lds_b128((gvec128*)g, (lvec128*)l, 0, 0);
#else
    *(u32x4*)l = *(const u32x4*)g;
#endif
}

__device__ __forceinline__ void wait_async0() {
#if HAVE_ASYNC_LDS
#  if __has_builtin(__builtin_amdgcn_s_wait_asynccnt)
    __builtin_amdgcn_s_wait_asynccnt(0);
#  else
    asm volatile("s_wait_asynccnt 0x0" ::: "memory");
#  endif
#endif
}

__device__ __forceinline__ unsigned short f2bf(float f) {
    unsigned int u = __builtin_bit_cast(unsigned int, f);
    u += 0x7fffu + ((u >> 16) & 1u);            // round-to-nearest-even
    return (unsigned short)(u >> 16);
}

// ------------------------------- init --------------------------------------
__global__ void moe_init_kernel(int* counts) {
    if (threadIdx.x < NEXP) counts[threadIdx.x] = 0;
}

// --------------------------- f32 -> bf16 -----------------------------------
__global__ void moe_cvt_bf16_kernel(const float* __restrict__ src,
                                    unsigned short* __restrict__ dst, int n) {
    int i = blockIdx.x * blockDim.x + threadIdx.x;
    int stride = gridDim.x * blockDim.x;
    for (; i < n; i += stride) dst[i] = f2bf(src[i]);
}

// ------------------------------ gating -------------------------------------
__global__ __launch_bounds__(256)
void moe_gate_kernel(const float* __restrict__ x, const float* __restrict__ wg,
                     int* __restrict__ counts,
                     int* __restrict__ expert_tokens, float* __restrict__ expert_gate,
                     int* __restrict__ topk_eid, int* __restrict__ topk_slot,
                     float* __restrict__ topk_gate) {
    const int wave = threadIdx.x >> 5;
    const int lane = threadIdx.x & 31;
    const int n = blockIdx.x * 8 + wave;
    if (n >= N_TOK) return;

    float acc[NEXP];
#pragma unroll
    for (int e = 0; e < NEXP; ++e) acc[e] = 0.0f;
    for (int k = lane; k < DIM; k += 32) {
        float xv = x[(size_t)n * DIM + k];
#pragma unroll
        for (int e = 0; e < NEXP; ++e) acc[e] += xv * wg[k * NEXP + e];
    }
#pragma unroll
    for (int e = 0; e < NEXP; ++e) {
        float v = acc[e];
        for (int off = 16; off >= 1; off >>= 1) v += __shfl_xor(v, off, 32);
        acc[e] = v;
    }
    if (lane == 0) {
        bool used[NEXP];
#pragma unroll
        for (int e = 0; e < NEXP; ++e) used[e] = false;
        float tg[TOPK]; int te[TOPK];
#pragma unroll
        for (int j = 0; j < TOPK; ++j) {                   // descending, first-index ties
            float best = -3.4e38f; int bi = 0;
            for (int e = 0; e < NEXP; ++e)
                if (!used[e] && acc[e] > best) { best = acc[e]; bi = e; }
            used[bi] = true; tg[j] = best; te[j] = bi;
        }
        float mx = tg[0], se = 0.0f, ex[TOPK];
#pragma unroll
        for (int j = 0; j < TOPK; ++j) { ex[j] = __expf(tg[j] - mx); se += ex[j]; }
#pragma unroll
        for (int j = 0; j < TOPK; ++j) {
            float g = ex[j] / se;
            int e = te[j];
            int s = atomicAdd(&counts[e], 1);
            expert_tokens[e * N_TOK + s] = n;
            expert_gate[e * N_TOK + s]  = g;
            topk_eid[n * TOPK + j]  = e;
            topk_slot[n * TOPK + j] = s;
            topk_gate[n * TOPK + j] = g;
        }
    }
}

// ----------------------------- offsets -------------------------------------
__global__ void moe_offsets_kernel(const int* __restrict__ counts, int* __restrict__ offsets) {
    if (threadIdx.x == 0) {
        int run = 0;
        for (int e = 0; e < NEXP; ++e) { offsets[e] = run; run += counts[e]; }
    }
}

// ------------------------------- loss --------------------------------------
__global__ __launch_bounds__(256)
void moe_loss_kernel(const float* __restrict__ topk_gate, const int* __restrict__ topk_eid,
                     const int* __restrict__ counts, float* __restrict__ loss_out) {
    __shared__ float red[256][NEXP];
    const int tid = threadIdx.x;
    float imp[NEXP];
#pragma unroll
    for (int e = 0; e < NEXP; ++e) imp[e] = 0.0f;
    for (int p = tid; p < N_TOK * TOPK; p += 256) imp[topk_eid[p]] += topk_gate[p];
#pragma unroll
    for (int e = 0; e < NEXP; ++e) red[tid][e] = imp[e];
    __syncthreads();
    for (int s = 128; s > 0; s >>= 1) {
        if (tid < s)
#pragma unroll
            for (int e = 0; e < NEXP; ++e) red[tid][e] += red[tid + s][e];
        __syncthreads();
    }
    if (tid == 0) {
        float cv[2];
        for (int which = 0; which < 2; ++which) {
            float v[NEXP], mean = 0.0f;
            for (int e = 0; e < NEXP; ++e) {
                v[e] = (which == 0) ? red[0][e] : (float)counts[e];
                mean += v[e];
            }
            mean *= (1.0f / NEXP);
            float var = 0.0f;
            for (int e = 0; e < NEXP; ++e) { float d = v[e] - mean; var += d * d; }
            var *= (1.0f / (NEXP - 1));                 // ddof=1
            cv[which] = var / (mean * mean + 1e-10f);
        }
        loss_out[0] = 0.01f * (cv[0] + cv[1]);
    }
}

// --------------------------- GEMM1: H = gelu(X @ W1) ------------------------
// 128x128 tile / WG; 8 waves as 2(M) x 4(N); wave = 64x32 = 4x2 WMMA frags.
// Double-buffered LDS, async DMA staging, one barrier per 64-wide K step.
__global__ __launch_bounds__(256)
void moe_gemm1_kernel(const unsigned short* __restrict__ xb,
                      const unsigned short* __restrict__ W1b,
                      const float* __restrict__ b1,
                      const int* __restrict__ expert_tokens,
                      const int* __restrict__ counts, const int* __restrict__ offsets,
                      unsigned short* __restrict__ H) {
    const int e   = blockIdx.z;
    const int cnt = counts[e];
    const int m0  = blockIdx.y * 128;
    if (m0 >= cnt) return;
    const int n0  = blockIdx.x * 128;
    const unsigned short* W = W1b + (size_t)e * DIM * FFN;

    __shared__ unsigned short Alds[2][128 * LDSW];
    __shared__ unsigned short Blds[2][128 * LDSW];

    const int tid  = threadIdx.x;
    const int lane = tid & 31;
    const int wave = tid >> 5;
    const int wm   = (wave & 1) * 64;
    const int wn   = (wave >> 1) * 32;
    const int lr   = lane & 15;
    const int lk   = (lane >> 4) * 8;      // K base: 0 (lanes 0-15) or 8 (16-31)

    v8f acc[4][2];
    const v8f vzero = {0.f, 0.f, 0.f, 0.f, 0.f, 0.f, 0.f, 0.f};
#pragma unroll
    for (int mi = 0; mi < 4; ++mi)
#pragma unroll
        for (int ni = 0; ni < 2; ++ni) acc[mi][ni] = vzero;

    auto stage = [&](int buf, int k0) {
        // A: 128 rows x 64 K, gathered by routed token, async DMA to LDS
#pragma unroll
        for (int it = 0; it < 4; ++it) {
            int chunk = tid + it * 256;
            int r = chunk >> 3, c8 = (chunk & 7) * 8;
            int row = m0 + r; if (row >= cnt) row = cnt - 1;
            int tok = expert_tokens[e * N_TOK + row];
            g2l_b128(xb + (size_t)tok * DIM + k0 + c8, &Alds[buf][r * LDSW + c8]);
        }
        // B: 64 K rows x 128 N, transpose-scatter into Blds[n][k]
#pragma unroll
        for (int it = 0; it < 4; ++it) {
            int chunk = tid + it * 256;
            int kr = chunk >> 4, c8 = (chunk & 15) * 8;
            const unsigned short* src = W + (size_t)(k0 + kr) * FFN + n0 + c8;
            if (k0 + KSTEP < DIM)                       // prefetch next K tile (L2 hit path)
                __builtin_prefetch(src + (size_t)KSTEP * FFN, 0, 1);
            Vec8 tv; tv.q = *(const u32x4*)src;
#pragma unroll
            for (int j = 0; j < 8; ++j) Blds[buf][(c8 + j) * LDSW + kr] = tv.s[j];
        }
    };

    auto compute = [&](int buf) {
#pragma unroll
        for (int sub = 0; sub < 2; ++sub) {
            const int kb = sub * 32 + lk;
            FragAB a[4], b[2];
#pragma unroll
            for (int f = 0; f < 4; ++f) {
                const unsigned short* p = &Alds[buf][(wm + f * 16 + lr) * LDSW + kb];
                a[f].q[0] = *(const u32x4*)p;
                a[f].q[1] = *(const u32x4*)(p + 16);
            }
#pragma unroll
            for (int f = 0; f < 2; ++f) {
                const unsigned short* p = &Blds[buf][(wn + f * 16 + lr) * LDSW + kb];
                b[f].q[0] = *(const u32x4*)p;
                b[f].q[1] = *(const u32x4*)(p + 16);
            }
#pragma unroll
            for (int mi = 0; mi < 4; ++mi)
#pragma unroll
                for (int ni = 0; ni < 2; ++ni)
                    acc[mi][ni] = __builtin_amdgcn_wmma_f32_16x16x32_bf16(
                        false, a[mi].f, false, b[ni].f, (short)0, acc[mi][ni], false, false);
        }
    };

    const int KT = DIM / KSTEP;
    stage(0, 0);
    wait_async0();
    __syncthreads();
    for (int kt = 0; kt < KT; ++kt) {
        int p = kt & 1;
        if (kt + 1 < KT) stage(1 - p, (kt + 1) * KSTEP);
        compute(p);
        if (kt + 1 < KT) { wait_async0(); __syncthreads(); }
    }

    // Epilogue: +b1, gelu(tanh approx), bf16 store to H
    const int hbase = offsets[e];
#pragma unroll
    for (int mi = 0; mi < 4; ++mi)
#pragma unroll
        for (int ni = 0; ni < 2; ++ni)
#pragma unroll
            for (int r = 0; r < 8; ++r) {
                int s = m0 + wm + mi * 16 + r + (lane >> 4) * 8;
                int n = n0 + wn + ni * 16 + (lane & 15);
                if (s < cnt) {
                    float v = acc[mi][ni][r] + b1[e * FFN + n];
                    float t = tanhf(0.7978845608028654f * (v + 0.044715f * v * v * v));
                    float g = 0.5f * v * (1.0f + t);
                    H[(size_t)(hbase + s) * FFN + n] = f2bf(g);
                }
            }
}

// ------------------- GEMM2: out = gate * exp(H @ W2 + b2) -------------------
__global__ __launch_bounds__(256)
void moe_gemm2_kernel(const unsigned short* __restrict__ H,
                      const unsigned short* __restrict__ W2b,
                      const float* __restrict__ b2,
                      const float* __restrict__ expert_gate,
                      const int* __restrict__ counts, const int* __restrict__ offsets,
                      float* __restrict__ outb) {
    const int e   = blockIdx.z;
    const int cnt = counts[e];
    const int m0  = blockIdx.y * 128;
    if (m0 >= cnt) return;
    const int n0    = blockIdx.x * 128;
    const int hbase = offsets[e];
    const unsigned short* W = W2b + (size_t)e * FFN * DIM;

    __shared__ unsigned short Alds[2][128 * LDSW];
    __shared__ unsigned short Blds[2][128 * LDSW];

    const int tid  = threadIdx.x;
    const int lane = tid & 31;
    const int wave = tid >> 5;
    const int wm   = (wave & 1) * 64;
    const int wn   = (wave >> 1) * 32;
    const int lr   = lane & 15;
    const int lk   = (lane >> 4) * 8;

    v8f acc[4][2];
    const v8f vzero = {0.f, 0.f, 0.f, 0.f, 0.f, 0.f, 0.f, 0.f};
#pragma unroll
    for (int mi = 0; mi < 4; ++mi)
#pragma unroll
        for (int ni = 0; ni < 2; ++ni) acc[mi][ni] = vzero;

    auto stage = [&](int buf, int k0) {
#pragma unroll
        for (int it = 0; it < 4; ++it) {
            int chunk = tid + it * 256;
            int r = chunk >> 3, c8 = (chunk & 7) * 8;
            int row = m0 + r; if (row >= cnt) row = cnt - 1;
            g2l_b128(H + (size_t)(hbase + row) * FFN + k0 + c8, &Alds[buf][r * LDSW + c8]);
        }
#pragma unroll
        for (int it = 0; it < 4; ++it) {
            int chunk = tid + it * 256;
            int kr = chunk >> 4, c8 = (chunk & 15) * 8;
            const unsigned short* src = W + (size_t)(k0 + kr) * DIM + n0 + c8;
            if (k0 + KSTEP < FFN)
                __builtin_prefetch(src + (size_t)KSTEP * DIM, 0, 1);
            Vec8 tv; tv.q = *(const u32x4*)src;
#pragma unroll
            for (int j = 0; j < 8; ++j) Blds[buf][(c8 + j) * LDSW + kr] = tv.s[j];
        }
    };

    auto compute = [&](int buf) {
#pragma unroll
        for (int sub = 0; sub < 2; ++sub) {
            const int kb = sub * 32 + lk;
            FragAB a[4], b[2];
#pragma unroll
            for (int f = 0; f < 4; ++f) {
                const unsigned short* p = &Alds[buf][(wm + f * 16 + lr) * LDSW + kb];
                a[f].q[0] = *(const u32x4*)p;
                a[f].q[1] = *(const u32x4*)(p + 16);
            }
#pragma unroll
            for (int f = 0; f < 2; ++f) {
                const unsigned short* p = &Blds[buf][(wn + f * 16 + lr) * LDSW + kb];
                b[f].q[0] = *(const u32x4*)p;
                b[f].q[1] = *(const u32x4*)(p + 16);
            }
#pragma unroll
            for (int mi = 0; mi < 4; ++mi)
#pragma unroll
                for (int ni = 0; ni < 2; ++ni)
                    acc[mi][ni] = __builtin_amdgcn_wmma_f32_16x16x32_bf16(
                        false, a[mi].f, false, b[ni].f, (short)0, acc[mi][ni], false, false);
        }
    };

    const int KT = FFN / KSTEP;
    stage(0, 0);
    wait_async0();
    __syncthreads();
    for (int kt = 0; kt < KT; ++kt) {
        int p = kt & 1;
        if (kt + 1 < KT) stage(1 - p, (kt + 1) * KSTEP);
        compute(p);
        if (kt + 1 < KT) { wait_async0(); __syncthreads(); }
    }

#pragma unroll
    for (int mi = 0; mi < 4; ++mi)
#pragma unroll
        for (int ni = 0; ni < 2; ++ni)
#pragma unroll
            for (int r = 0; r < 8; ++r) {
                int s = m0 + wm + mi * 16 + r + (lane >> 4) * 8;
                int n = n0 + wn + ni * 16 + (lane & 15);
                if (s < cnt) {
                    float v = acc[mi][ni][r] + b2[e * DIM + n];
                    float g = expert_gate[e * N_TOK + s];
                    outb[(size_t)(hbase + s) * DIM + n] = g * expf(v);
                }
            }
}

// ------------------------------ combine ------------------------------------
__global__ __launch_bounds__(256)
void moe_combine_kernel(const float* __restrict__ outb,
                        const int* __restrict__ topk_eid, const int* __restrict__ topk_slot,
                        const int* __restrict__ offsets, float* __restrict__ y) {
    const int n = blockIdx.x;
    int row[TOPK];
#pragma unroll
    for (int j = 0; j < TOPK; ++j)
        row[j] = offsets[topk_eid[n * TOPK + j]] + topk_slot[n * TOPK + j];
    for (int d = threadIdx.x; d < DIM; d += 256) {
        float s = 0.0f;
#pragma unroll
        for (int j = 0; j < TOPK; ++j) s += outb[(size_t)row[j] * DIM + d];
        if (s == 0.0f) s = 2.220446049250313e-16f;   // np.finfo(float).eps
        y[(size_t)n * DIM + d] = logf(s);
    }
}

// ------------------------------- launch ------------------------------------
extern "C" void kernel_launch(void* const* d_in, const int* in_sizes, int n_in,
                              void* d_out, int out_size, void* d_ws, size_t ws_size,
                              hipStream_t stream) {
    const float* x      = (const float*)d_in[0];
    const float* w_gate = (const float*)d_in[1];
    const float* W1     = (const float*)d_in[2];
    const float* b1     = (const float*)d_in[3];
    const float* W2     = (const float*)d_in[4];
    const float* b2     = (const float*)d_in[5];
    float* y = (float*)d_out;                       // [N_TOK*DIM] + loss at end

    // Workspace layout (256B aligned): ~345 MB
    char* ws = (char*)d_ws;
    size_t off = 0;
    auto take = [&](size_t bytes) { char* p = ws + off; off = (off + bytes + 255) & ~(size_t)255; return p; };
    unsigned short* xb  = (unsigned short*)take((size_t)N_TOK * DIM * 2);
    unsigned short* W1b = (unsigned short*)take((size_t)NEXP * DIM * FFN * 2);
    unsigned short* W2b = (unsigned short*)take((size_t)NEXP * FFN * DIM * 2);
    unsigned short* H   = (unsigned short*)take((size_t)N_TOK * TOPK * FFN * 2);
    float* outb         = (float*)take((size_t)N_TOK * TOPK * DIM * 4);
    int*   expert_tokens= (int*)take((size_t)NEXP * N_TOK * 4);
    float* expert_gate  = (float*)take((size_t)NEXP * N_TOK * 4);
    int*   topk_eid     = (int*)take((size_t)N_TOK * TOPK * 4);
    int*   topk_slot    = (int*)take((size_t)N_TOK * TOPK * 4);
    float* topk_gate    = (float*)take((size_t)N_TOK * TOPK * 4);
    int*   counts       = (int*)take(NEXP * 4);
    int*   offsets      = (int*)take(NEXP * 4);
    (void)in_sizes; (void)n_in; (void)out_size; (void)ws_size;

    moe_init_kernel<<<1, 32, 0, stream>>>(counts);

    moe_cvt_bf16_kernel<<<2048, 256, 0, stream>>>(x,  xb,  N_TOK * DIM);
    moe_cvt_bf16_kernel<<<4096, 256, 0, stream>>>(W1, W1b, NEXP * DIM * FFN);
    moe_cvt_bf16_kernel<<<4096, 256, 0, stream>>>(W2, W2b, NEXP * FFN * DIM);

    moe_gate_kernel<<<N_TOK / 8, 256, 0, stream>>>(x, w_gate, counts, expert_tokens,
                                                   expert_gate, topk_eid, topk_slot, topk_gate);
    moe_offsets_kernel<<<1, 32, 0, stream>>>(counts, offsets);
    moe_loss_kernel<<<1, 256, 0, stream>>>(topk_gate, topk_eid, counts, y + (size_t)N_TOK * DIM);

    dim3 g1(FFN / 128, N_TOK / 128, NEXP);          // blocks beyond counts[e] early-exit
    moe_gemm1_kernel<<<g1, 256, 0, stream>>>(xb, W1b, b1, expert_tokens, counts, offsets, H);

    dim3 g2(DIM / 128, N_TOK / 128, NEXP);
    moe_gemm2_kernel<<<g2, 256, 0, stream>>>(H, W2b, b2, expert_gate, counts, offsets, outb);

    moe_combine_kernel<<<N_TOK, 256, 0, stream>>>(outb, topk_eid, topk_slot, offsets, y);
}
